// AttentionBridge_22591527977644
// MI455X (gfx1250) — matrix operand order repaired and checked by reading, not compiled, not run on hardware
//
#include <hip/hip_runtime.h>
#include <hip/hip_bf16.h>

// ---------------------------------------------------------------------------
// AttentionBridge on MI455X (gfx1250, wave32, WMMA bf16 w/ f32 accumulation)
// Pipeline: cast->bf16, GEMM(Q,K,V proj; V stored head-transposed), flash
// attention with ALiBi, GEMM(out proj + residual), RMSNorm.
// ---------------------------------------------------------------------------

#define D_MODEL 1024
#define NUM_HEADS 16
#define HEAD_DIM 64
#define T_LEN 1024
#define S_LEN 2048
#define BATCH 2
#define QK_SCALE 0.125f // 64^-0.5
#define RMS_EPS 1e-6f

typedef __attribute__((ext_vector_type(8)))  unsigned short us8;
typedef __attribute__((ext_vector_type(16))) unsigned short us16;
typedef __attribute__((ext_vector_type(16))) __bf16        v16bf;
typedef __attribute__((ext_vector_type(8)))  float         v8f;

__device__ __forceinline__ unsigned short f2bf(float f) {
    unsigned int u = __float_as_uint(f);
    unsigned int r = u + 0x7FFFu + ((u >> 16) & 1u); // round-to-nearest-even
    return (unsigned short)(r >> 16);
}

// A-fragment (16x32, MxK): lane holds row (L&15), K = 16*(L>>4)+{0..15}
// -> one contiguous 32B chunk at p (p already offset by 16*(L>>4)).
__device__ __forceinline__ v16bf make_afrag(const unsigned short* p) {
    us16 t = *(const us16*)(p);
    return __builtin_bit_cast(v16bf, t);
}

// B-fragment (32x16, KxN): lane holds col (L&15), K = 16*(L>>4)+{0..15}
// -> one contiguous 32B chunk when the source is stored as B^T row-major.
__device__ __forceinline__ v16bf load_bfrag(const unsigned short* p) {
    return *(const v16bf*)(p);
}

__device__ __forceinline__ v8f wmma_bf16(v16bf a, v16bf b, v8f c) {
    return __builtin_amdgcn_wmma_f32_16x16x32_bf16(false, a, false, b, (short)0, c,
                                                   false, false);
}

// v_mov_b32_dpp row rotate (within the 16-lane DPP row; wave32 = 2 rows).
// Combining with ror 1,2,4,8 gives every lane the full 16-lane reduction,
// entirely in the VALU (no ds_bpermute / LDS traffic).
template <int CTRL>
__device__ __forceinline__ float dpp_rot(float x) {
    int r = __builtin_amdgcn_update_dpp(0, __float_as_int(x), CTRL, 0xF, 0xF, true);
    return __int_as_float(r);
}
__device__ __forceinline__ float row16_max(float x) {
    x = fmaxf(x, dpp_rot<0x121>(x)); // row_ror:1
    x = fmaxf(x, dpp_rot<0x122>(x)); // row_ror:2
    x = fmaxf(x, dpp_rot<0x124>(x)); // row_ror:4
    x = fmaxf(x, dpp_rot<0x128>(x)); // row_ror:8
    return x;
}
__device__ __forceinline__ float row16_sum(float x) {
    x += dpp_rot<0x121>(x);
    x += dpp_rot<0x122>(x);
    x += dpp_rot<0x124>(x);
    x += dpp_rot<0x128>(x);
    return x;
}

// ---------------------------------------------------------------------------
__global__ void cast_bf16_kernel(const float* __restrict__ src,
                                 unsigned short* __restrict__ dst, int n) {
    int i = blockIdx.x * blockDim.x + threadIdx.x;
    if (i < n) dst[i] = f2bf(src[i]);
}

// ---------------------------------------------------------------------------
// C[M,N] = A[M,K] @ Bt[N,K]^T   (i.e. x @ W.T with Bt = W row-major)
// Block = 64x64 tile, 4 waves; wave computes a 16x64 strip (4 accumulators).
// MODE 0: store bf16 row-major.  MODE 1: store f32 + residual add.
// MODE 2: store bf16 head-transposed  Vt[b][h][d][s]  (V projection).
template <int MODE>
__global__ __launch_bounds__(128) void gemm_bf16_kernel(
    const unsigned short* __restrict__ A, const unsigned short* __restrict__ Bt,
    void* __restrict__ Cout, const float* __restrict__ resid,
    int M, int N, int K) {
    const int lane = threadIdx.x & 31;
    const int wave = threadIdx.x >> 5;
    const int lrow = lane & 15;
    const int hi   = lane >> 4;
    const int m0   = blockIdx.y * 64 + wave * 16;
    const int n0   = blockIdx.x * 64;

    const unsigned short* arow = A  + (size_t)(m0 + lrow) * K + 16 * hi;
    const unsigned short* brow = Bt + (size_t)(n0 + lrow) * K + 16 * hi;

    v8f acc[4] = {v8f{}, v8f{}, v8f{}, v8f{}};
    for (int k0 = 0; k0 < K; k0 += 32) {
        if (k0 + 32 < K) {
            __builtin_prefetch(arow + k0 + 32, 0, 1);
            __builtin_prefetch(brow + k0 + 32, 0, 1);
        }
        v16bf a = make_afrag(arow + k0);
#pragma unroll
        for (int j = 0; j < 4; ++j) {
            v16bf b = load_bfrag(brow + (size_t)j * 16 * K + k0);
            acc[j] = wmma_bf16(a, b, acc[j]);
        }
    }

#pragma unroll
    for (int r = 0; r < 8; ++r) {
        const int m = m0 + r + 8 * hi;
#pragma unroll
        for (int j = 0; j < 4; ++j) {
            const int n = n0 + j * 16 + lrow;
            const float v = acc[j][r];
            if (MODE == 0) {
                ((unsigned short*)Cout)[(size_t)m * N + n] = f2bf(v);
            } else if (MODE == 1) {
                ((float*)Cout)[(size_t)m * N + n] = v + resid[(size_t)m * N + n];
            } else { // MODE 2: m = b*S + s ; n = h*64 + d ; Vt[b][h][d][s]
                const int b = m >> 11, s = m & (S_LEN - 1);
                const int h = n >> 6, d = n & 63;
                ((unsigned short*)Cout)[(((size_t)b * NUM_HEADS + h) * HEAD_DIM + d)
                                        * S_LEN + s] = f2bf(v);
            }
        }
    }
}

// ---------------------------------------------------------------------------
// Flash attention with ALiBi. Block = 4 waves; wave owns 16 query rows of one
// (b,h). s-chunks of 32: scores via 4 WMMAs, online softmax (DPP row-rotate
// all-reduce, no LDS), P re-laid out via LDS round trip into an A-fragment,
// then 4 P·V WMMAs against transposed V. ALiBi distance tracked in float regs.
__global__ __launch_bounds__(128) void attn_kernel(
    const unsigned short* __restrict__ Qb, const unsigned short* __restrict__ Kb,
    const unsigned short* __restrict__ Vt, unsigned short* __restrict__ Ab) {
    __shared__ __align__(32) unsigned short pbuf[4][16][32];

    const int lane = threadIdx.x & 31;
    const int wave = threadIdx.x >> 5;
    const int lrow = lane & 15;
    const int hi   = lane >> 4;
    const int b    = blockIdx.y >> 4;
    const int h    = blockIdx.y & 15;
    const int t0   = blockIdx.x * 64 + wave * 16;

    const float slope = exp2f(-(float)(h + 1) * (1.0f / 16.0f));

    // Q fragments (hd split into two K=32 halves), kept in registers.
    const unsigned short* qrow =
        Qb + (size_t)(b * T_LEN + t0 + lrow) * D_MODEL + h * HEAD_DIM + 16 * hi;
    const v16bf qa0 = make_afrag(qrow);
    const v16bf qa1 = make_afrag(qrow + 32);

    const unsigned short* kbase =
        Kb + (size_t)(b * S_LEN + lrow) * D_MODEL + h * HEAD_DIM + 16 * hi;
    const unsigned short* vbase =
        Vt + ((size_t)(b * NUM_HEADS + h) * HEAD_DIM) * S_LEN;

    float rm[8], rs[8], tdist[8];
#pragma unroll
    for (int r = 0; r < 8; ++r) {
        rm[r] = -1e30f;
        rs[r] = 0.0f;
        tdist[r] = (float)(t0 + r + 8 * hi - lrow); // (t - s_lane) at s0 == 0
    }
    v8f oc[4] = {v8f{}, v8f{}, v8f{}, v8f{}};

    float sbase = 0.0f; // float mirror of s0 (exact: integers < 2^24)
    for (int s0 = 0; s0 < S_LEN; s0 += 32) {
        // scores for s-tiles [s0, s0+16) and [s0+16, s0+32)
        v8f sc0 = {}, sc1 = {};
        sc0 = wmma_bf16(qa0, load_bfrag(kbase + (size_t)s0 * D_MODEL), sc0);
        sc0 = wmma_bf16(qa1, load_bfrag(kbase + (size_t)s0 * D_MODEL + 32), sc0);
        sc1 = wmma_bf16(qa0, load_bfrag(kbase + (size_t)(s0 + 16) * D_MODEL), sc1);
        sc1 = wmma_bf16(qa1, load_bfrag(kbase + (size_t)(s0 + 16) * D_MODEL + 32), sc1);

#pragma unroll
        for (int r = 0; r < 8; ++r) {
            const float da = tdist[r] - sbase;          // t - s  (tile 0)
            const float db = da - 16.0f;                // t - s  (tile 1)
            float sa = sc0[r] * QK_SCALE - slope * fabsf(da);
            float sb = sc1[r] * QK_SCALE - slope * fabsf(db);
            // row max/sum across the 16 lanes holding this row (DPP rotate)
            const float mx   = row16_max(fmaxf(sa, sb));
            const float nm   = fmaxf(rm[r], mx);
            const float corr = __expf(rm[r] - nm);
            const float ea   = __expf(sa - nm);
            const float eb   = __expf(sb - nm);
            const float sum  = row16_sum(ea + eb);
            rs[r] = rs[r] * corr + sum;
            rm[r] = nm;
#pragma unroll
            for (int j = 0; j < 4; ++j) oc[j][r] *= corr;
            // stage P (C-layout) to LDS for re-read in A-fragment layout
            pbuf[wave][r + 8 * hi][lrow]      = f2bf(ea);
            pbuf[wave][r + 8 * hi][16 + lrow] = f2bf(eb);
        }
        asm volatile("s_wait_dscnt 0" ::: "memory"); // wave-private LDS RAW

        const v16bf pf = make_afrag(&pbuf[wave][lrow][16 * hi]);
#pragma unroll
        for (int j = 0; j < 4; ++j) {
            v16bf vf = load_bfrag(vbase + (size_t)(j * 16 + lrow) * S_LEN
                                  + s0 + 16 * hi);
            oc[j] = wmma_bf16(pf, vf, oc[j]);
        }
        sbase += 32.0f;
    }

#pragma unroll
    for (int r = 0; r < 8; ++r) {
        const float inv = 1.0f / rs[r];
        const size_t row = (size_t)(b * T_LEN + t0 + r + 8 * hi) * D_MODEL;
#pragma unroll
        for (int j = 0; j < 4; ++j)
            Ab[row + h * HEAD_DIM + j * 16 + lrow] = f2bf(oc[j][r] * inv);
    }
}

// ---------------------------------------------------------------------------
__global__ __launch_bounds__(256) void rmsnorm_kernel(
    const float* __restrict__ Z, const float* __restrict__ w,
    float* __restrict__ out) {
    __shared__ float red[256];
    const int row = blockIdx.x;
    const float* z = Z + (size_t)row * D_MODEL;
    float acc = 0.0f;
    for (int i = threadIdx.x; i < D_MODEL; i += 256) { float v = z[i]; acc += v * v; }
    red[threadIdx.x] = acc;
    __syncthreads();
    for (int s = 128; s > 0; s >>= 1) {
        if (threadIdx.x < s) red[threadIdx.x] += red[threadIdx.x + s];
        __syncthreads();
    }
    const float sc = rsqrtf(red[0] * (1.0f / D_MODEL) + RMS_EPS);
    for (int i = threadIdx.x; i < D_MODEL; i += 256)
        out[(size_t)row * D_MODEL + i] = z[i] * sc * w[i];
}

// ---------------------------------------------------------------------------
extern "C" void kernel_launch(void* const* d_in, const int* in_sizes, int n_in,
                              void* d_out, int out_size, void* d_ws, size_t ws_size,
                              hipStream_t stream) {
    (void)in_sizes; (void)n_in; (void)out_size; (void)ws_size;
    const float* query   = (const float*)d_in[0]; // [B*T, D]
    const float* context = (const float*)d_in[1]; // [B*S, D]
    const float* Wq      = (const float*)d_in[2];
    const float* Wk      = (const float*)d_in[3];
    const float* Wv      = (const float*)d_in[4];
    const float* Wo      = (const float*)d_in[5];
    const float* rms_w   = (const float*)d_in[6];

    const int MQ = BATCH * T_LEN;  // 2048
    const int MK = BATCH * S_LEN;  // 4096
    const size_t MB = 1ull << 20;

    char* ws = (char*)d_ws;        // 52 MB total
    unsigned short* Xq  = (unsigned short*)(ws + 0 * MB);   // 4 MB
    unsigned short* Xc  = (unsigned short*)(ws + 4 * MB);   // 8 MB
    unsigned short* Wqb = (unsigned short*)(ws + 12 * MB);  // 2 MB
    unsigned short* Wkb = (unsigned short*)(ws + 14 * MB);  // 2 MB
    unsigned short* Wvb = (unsigned short*)(ws + 16 * MB);  // 2 MB
    unsigned short* Wob = (unsigned short*)(ws + 18 * MB);  // 2 MB
    unsigned short* Qb  = (unsigned short*)(ws + 20 * MB);  // 4 MB
    unsigned short* Kb  = (unsigned short*)(ws + 24 * MB);  // 8 MB
    unsigned short* Vtb = (unsigned short*)(ws + 32 * MB);  // 8 MB (transposed)
    unsigned short* Ab  = (unsigned short*)(ws + 40 * MB);  // 4 MB
    float*          Z   = (float*)(ws + 44 * MB);           // 8 MB

    const int ct = 256;
    cast_bf16_kernel<<<(MQ * D_MODEL + ct - 1) / ct, ct, 0, stream>>>(query, Xq, MQ * D_MODEL);
    cast_bf16_kernel<<<(MK * D_MODEL + ct - 1) / ct, ct, 0, stream>>>(context, Xc, MK * D_MODEL);
    cast_bf16_kernel<<<(D_MODEL * D_MODEL + ct - 1) / ct, ct, 0, stream>>>(Wq, Wqb, D_MODEL * D_MODEL);
    cast_bf16_kernel<<<(D_MODEL * D_MODEL + ct - 1) / ct, ct, 0, stream>>>(Wk, Wkb, D_MODEL * D_MODEL);
    cast_bf16_kernel<<<(D_MODEL * D_MODEL + ct - 1) / ct, ct, 0, stream>>>(Wv, Wvb, D_MODEL * D_MODEL);
    cast_bf16_kernel<<<(D_MODEL * D_MODEL + ct - 1) / ct, ct, 0, stream>>>(Wo, Wob, D_MODEL * D_MODEL);

    dim3 gq(D_MODEL / 64, MQ / 64), gk(D_MODEL / 64, MK / 64);
    gemm_bf16_kernel<0><<<gq, 128, 0, stream>>>(Xq, Wqb, Qb, nullptr, MQ, D_MODEL, D_MODEL);
    gemm_bf16_kernel<0><<<gk, 128, 0, stream>>>(Xc, Wkb, Kb, nullptr, MK, D_MODEL, D_MODEL);
    gemm_bf16_kernel<2><<<gk, 128, 0, stream>>>(Xc, Wvb, Vtb, nullptr, MK, D_MODEL, D_MODEL);

    attn_kernel<<<dim3(T_LEN / 64, BATCH * NUM_HEADS), 128, 0, stream>>>(Qb, Kb, Vtb, Ab);

    gemm_bf16_kernel<1><<<gq, 128, 0, stream>>>(Ab, Wob, Z, query, MQ, D_MODEL, D_MODEL);
    rmsnorm_kernel<<<MQ, 256, 0, stream>>>(Z, rms_w, (float*)d_out);
}
